// TreeEnergyLoss_22514218566396
// MI455X (gfx1250) — compile-verified, matrix operand order checked by reading, not run on hardware
//
#include <hip/hip_runtime.h>
#include <hip/hip_bf16.h>

// ---------------- problem constants (from setup_inputs) ----------------
#define B_    8
#define HH    80
#define WW    80
#define V_    6400            // H*W
#define EV_   6320            // vertical edges (H-1)*W
#define E_    12640           // total 4-connected edges
#define C_    21              // prediction channels
#define NSORT 16384           // edges padded to pow2 for bitonic sort
#define SIGMA_ 0.002f
#define CPG   4               // channels per tree-filter group
#define NGRP  6               // ceil((C_+1)/CPG) channel groups

typedef unsigned long long u64;
typedef __attribute__((ext_vector_type(16))) __bf16 v16bf;
typedef __attribute__((ext_vector_type(8)))  float  v8f;

// ---------------- helpers ----------------
__device__ __forceinline__ void edge_uv(int e, int& u, int& v) {
    if (e < EV_) { u = e; v = e + WW; }                 // vertical: idx[:-1,:] -> idx[1:,:]
    else {
        int t = e - EV_;
        int r = t / (WW - 1), c = t % (WW - 1);
        u = r * WW + c; v = u + 1;                      // horizontal: idx[:,:-1] -> idx[:,1:]
    }
}

// ---------------- softmax over channel axis ----------------
__global__ void softmax_k(const float* __restrict__ preds, float* __restrict__ prob) {
    int t = blockIdx.x * blockDim.x + threadIdx.x;
    if (t >= B_ * V_) return;
    int b = t / V_, v = t % V_;
    const float* p = preds + (size_t)b * C_ * V_ + v;
    float m = -1e30f;
    for (int c = 0; c < C_; ++c) m = fmaxf(m, p[(size_t)c * V_]);
    float e[C_], s = 0.f;
    for (int c = 0; c < C_; ++c) { e[c] = expf(p[(size_t)c * V_] - m); s += e[c]; }
    float inv = 1.f / s;
    float* o = prob + (size_t)b * C_ * V_ + v;
    for (int c = 0; c < C_; ++c) o[(size_t)c * V_] = e[c] * inv;
}

// ---------------- MST (Kruskal, stable by (cost,idx)) + BFS ordering ----------------
// One workgroup per batch. Edge keys sorted with a 128KB LDS bitonic sort
// (CDNA5: 320KB LDS per WGP). Union-find / adjacency / BFS are tiny & serial.
__global__ void __launch_bounds__(1024) build_tree_k(
    const float* __restrict__ embed, int CE,
    int* __restrict__ si_all, int* __restrict__ sp_all,
    int* __restrict__ ufp_all, int* __restrict__ mstu_all, int* __restrict__ mstv_all,
    int* __restrict__ deg_all, int* __restrict__ off_all,
    int* __restrict__ adj_all, int* __restrict__ vis_all)
{
    extern __shared__ char dynlds[];
    u64* keys = (u64*)dynlds;                           // NSORT * 8B = 128KB LDS
    const int b = blockIdx.x;
    const int tid = threadIdx.x;
    const float* emb = embed + (size_t)b * CE * V_;

    // edge costs -> sort keys (cost bits are >=0 floats: monotonic as u32; idx breaks ties = stable)
    for (int i = tid; i < NSORT; i += blockDim.x) {
        if (i < E_) {
            int u, v; edge_uv(i, u, v);
            float cost = 0.f;
            for (int c = 0; c < CE; ++c) {
                float d = emb[(size_t)c * V_ + u] - emb[(size_t)c * V_ + v];
                cost = fmaf(d, d, cost);
            }
            keys[i] = ((u64)__float_as_uint(cost) << 32) | (unsigned)i;
        } else {
            keys[i] = ~0ull;
        }
    }
    __syncthreads();

    // block-wide bitonic sort, ascending
    for (int k = 2; k <= NSORT; k <<= 1) {
        for (int j = k >> 1; j > 0; j >>= 1) {
            for (int i = tid; i < NSORT; i += blockDim.x) {
                int ij = i ^ j;
                if (ij > i) {
                    u64 a = keys[i], bb = keys[ij];
                    bool up = ((i & k) == 0);
                    if ((a > bb) == up) { keys[i] = bb; keys[ij] = a; }
                }
            }
            __syncthreads();
        }
    }

    if (tid == 0) {
        int* ufp  = ufp_all  + b * V_;
        int* mstu = mstu_all + b * (V_ - 1);
        int* mstv = mstv_all + b * (V_ - 1);
        int* deg  = deg_all  + b * V_;
        int* off  = off_all  + b * (V_ + 1);
        int* adj  = adj_all  + b * 2 * (V_ - 1);
        int* vis  = vis_all  + b * V_;
        int* si   = si_all   + b * V_;
        int* sp   = sp_all   + b * V_;

        for (int i = 0; i < V_; ++i) { ufp[i] = i; deg[i] = 0; vis[i] = 0; }

        // Kruskal with path compression, edges in sorted order
        int cnt = 0;
        for (int s = 0; s < E_ && cnt < V_ - 1; ++s) {
            int e = (int)(keys[s] & 0xffffffffu);
            int u, v; edge_uv(e, u, v);
            int ru = u; while (ufp[ru] != ru) ru = ufp[ru];
            int x = u;  while (ufp[x] != ru) { int nx = ufp[x]; ufp[x] = ru; x = nx; }
            int rv = v; while (ufp[rv] != rv) rv = ufp[rv];
            x = v;      while (ufp[x] != rv) { int nx = ufp[x]; ufp[x] = rv; x = nx; }
            if (ru != rv) {
                ufp[ru] = rv;
                mstu[cnt] = u; mstv[cnt] = v; ++cnt;
                deg[u]++; deg[v]++;
            }
        }
        // adjacency lists in edge-acceptance order (matches reference nbr append order)
        int acc = 0;
        for (int i = 0; i < V_; ++i) { off[i] = acc; acc += deg[i]; deg[i] = 0; }
        off[V_] = acc;
        for (int k2 = 0; k2 < cnt; ++k2) {
            int u = mstu[k2], v = mstv[k2];
            adj[off[u] + deg[u]++] = v;
            adj[off[v] + deg[v]++] = u;
        }
        // BFS from vertex 0; queue doubles as sorted_index
        si[0] = 0; sp[0] = 0; vis[0] = 1;
        int head = 0, tail = 1;
        while (head < tail) {
            int u = si[head];
            for (int a2 = off[u]; a2 < off[u + 1]; ++a2) {
                int v = adj[a2];
                if (!vis[v]) { vis[v] = 1; si[tail] = v; sp[tail] = head; ++tail; }
            }
            ++head;
        }
    }
}

// ---------------- per-node tree weights w[j] = exp(-||e[si[j]] - e[si[sp[j]]]||^2 / sigma) ----------------
__global__ void compute_w_k(const float* __restrict__ embed, int CE,
                            const int* __restrict__ si_all, const int* __restrict__ sp_all,
                            float* __restrict__ w_all) {
    int t = blockIdx.x * blockDim.x + threadIdx.x;
    if (t >= B_ * V_) return;
    int b = t / V_, j = t % V_;
    const int* si = si_all + b * V_;
    const int* sp = sp_all + b * V_;
    const float* emb = embed + (size_t)b * CE * V_;
    int u = si[j], p = si[sp[j]];
    float d = 0.f;
    for (int c = 0; c < CE; ++c) {
        float df = emb[(size_t)c * V_ + u] - emb[(size_t)c * V_ + p];
        d = fmaf(df, df, d);
    }
    w_all[b * V_ + j] = expf(-d / SIGMA_);
}

// ---------------- fused tree filter (up-scan + down-scan), fully LDS-resident ----------------
// Grid: (batch, channel-group). Per block: sp/si/w + Sc[CPG][V] + Ac[CPG][V] all in LDS
// (11*V*4 = 281.6KB <= 320KB/WGP on CDNA5). The 6399-step dependent chains run at LDS
// latency instead of HBM latency; intermediate S never touches global memory.
__global__ void __launch_bounds__(128) tree_filter_k(const float* __restrict__ f,
                                                     const int* __restrict__ si_all,
                                                     const int* __restrict__ sp_all,
                                                     const float* __restrict__ w_all,
                                                     float* __restrict__ A_all) {
    extern __shared__ char dynlds[];
    int*   sps = (int*)dynlds;                               // V
    int*   sis = (int*)(dynlds + (size_t)V_ * 4);            // V
    float* ws  = (float*)(dynlds + (size_t)V_ * 8);          // V
    float* Sc  = (float*)(dynlds + (size_t)V_ * 12);         // CPG * V
    float* Ac  = Sc + (size_t)CPG * V_;                      // CPG * V

    const int b   = blockIdx.x;
    const int c0  = blockIdx.y * CPG;                        // first channel of this group
    const int nch = min(CPG, (C_ + 1) - c0);
    const int tid = threadIdx.x, nt = blockDim.x;

    const int*   si = si_all + b * V_;
    const int*   sp = sp_all + b * V_;
    const float* w  = w_all  + b * V_;

    for (int j = tid; j < V_; j += nt) { sps[j] = sp[j]; sis[j] = si[j]; ws[j] = w[j]; }
    __syncthreads();
    for (int c = 0; c < nch; ++c) {
        const int ch = c0 + c;
        float* S = Sc + (size_t)c * V_;
        if (ch < C_) {
            const float* fc = f + (size_t)b * C_ * V_ + (size_t)ch * V_;
            for (int j = tid; j < V_; j += nt) S[j] = fc[sis[j]];
        } else {
            for (int j = tid; j < V_; j += nt) S[j] = 1.0f;   // normalizer channel
        }
    }
    __syncthreads();

    if (tid < nch) {
        float* S = Sc + (size_t)tid * V_;
        float* A = Ac + (size_t)tid * V_;
        // bottom-up: children (higher BFS pos) accumulate into parents
        for (int j = V_ - 1; j >= 1; --j) S[sps[j]] += ws[j] * S[j];
        // top-down: parents propagate to children
        A[0] = S[0];
        for (int j = 1; j < V_; ++j) {
            float wj = ws[j];
            A[j] = S[j] + wj * (A[sps[j]] - wj * S[j]);
        }
    }
    __syncthreads();

    for (int c = 0; c < nch; ++c) {
        float* Adst = A_all + (size_t)b * (C_ + 1) * V_ + (size_t)(c0 + c) * V_;
        const float* A = Ac + (size_t)c * V_;
        for (int j = tid; j < V_; j += nt) Adst[j] = A[j];
    }
}

// ---------------- normalize & scatter back to pixel order ----------------
__global__ void normalize_k(const float* __restrict__ A_all, const int* __restrict__ si_all,
                            float* __restrict__ out) {
    int t = blockIdx.x * blockDim.x + threadIdx.x;
    if (t >= B_ * C_ * V_) return;
    int b = t / (C_ * V_); int r = t % (C_ * V_); int c = r / V_; int j = r % V_;
    const float* A = A_all + (size_t)b * (C_ + 1) * V_;
    const int* si = si_all + b * V_;
    out[(size_t)b * C_ * V_ + (size_t)c * V_ + si[j]] = A[(size_t)c * V_ + j] / A[(size_t)C_ * V_ + j];
}

// ---------------- loss numerator via WMMA reduction ----------------
// D = A(16x32 bf16) x Ones(32x16) accumulated in f32: every packed element contributes
// exactly once to a row-sum independent of A's lane layout. TOT = 1050 * 1024 exactly,
// so waves iterate over whole 1024-element tiles with NO bounds checks. Each 16-element
// chunk is 16-aligned and V_ % 16 == 0, so it never crosses a v-wrap/batch boundary:
// the roi index is computed ONCE per chunk and roi reads are contiguous (vectorizable).
// Two accumulators break the WMMA->WMMA RAW chain. One partial per wave (deterministic).
__global__ void __launch_bounds__(256) loss_num_k(const float* __restrict__ prob,
                                                  const float* __restrict__ AS,
                                                  const int* __restrict__ roi,
                                                  float* __restrict__ partial) {
    const int TOT   = B_ * C_ * V_;       // 1,075,200
    const int NTILE = TOT / 1024;         // 1050 whole tiles, no remainder
    const int lane  = threadIdx.x & 31;
    const int wid   = (blockIdx.x * blockDim.x + threadIdx.x) >> 5;
    const int nw    = (gridDim.x * blockDim.x) >> 5;

    v8f cacc0 = {}, cacc1 = {};
    v16bf ones;
#pragma unroll
    for (int k = 0; k < 16; ++k) ones[k] = (__bf16)1.0f;

    for (int t = wid; t < NTILE; t += nw) {
        const int e0 = t * 1024 + lane * 16;              // 16-aligned
        const int e1 = e0 + 512;
        // roi base per chunk: chunk lies fully inside one (b, c) row of V elements
        const int bb0 = e0 / (C_ * V_);
        const int rb0 = bb0 * V_ + (e0 - bb0 * (C_ * V_)) % V_;
        const int bb1 = e1 / (C_ * V_);
        const int rb1 = bb1 * V_ + (e1 - bb1 * (C_ * V_)) % V_;
        // prefetch next sweep (speculative: OOB translations silently dropped)
        __builtin_prefetch(prob + e0 + nw * 1024, 0, 1);
        __builtin_prefetch(AS   + e0 + nw * 1024, 0, 1);

        v16bf a0, a1;
#pragma unroll
        for (int k = 0; k < 16; ++k)
            a0[k] = (__bf16)((float)roi[rb0 + k] * fabsf(prob[e0 + k] - AS[e0 + k]));
#pragma unroll
        for (int k = 0; k < 16; ++k)
            a1[k] = (__bf16)((float)roi[rb1 + k] * fabsf(prob[e1 + k] - AS[e1 + k]));

        cacc0 = __builtin_amdgcn_wmma_f32_16x16x32_bf16(false, a0, false, ones,
                                                        (short)0, cacc0, false, false);
        cacc1 = __builtin_amdgcn_wmma_f32_16x16x32_bf16(false, a1, false, ones,
                                                        (short)0, cacc1, false, false);
    }

    float local = 0.f;
#pragma unroll
    for (int i = 0; i < 8; ++i) local += cacc0[i] + cacc1[i];
    for (int off = 16; off > 0; off >>= 1) local += __shfl_down(local, off, 32);
    if (lane == 0) partial[wid] = local * 0.0625f;      // each rowsum replicated across 16 columns
}

// ---------------- loss denominator (ROI count), deterministic partials ----------------
__global__ void __launch_bounds__(256) loss_den_k(const int* __restrict__ roi, float* __restrict__ pden) {
    __shared__ float red[256];
    int t = blockIdx.x * blockDim.x + threadIdx.x;
    float s = 0.f;
    for (int i = t; i < B_ * V_; i += gridDim.x * blockDim.x) s += (float)roi[i];
    red[threadIdx.x] = s; __syncthreads();
    for (int o = 128; o > 0; o >>= 1) {
        if (threadIdx.x < o) red[threadIdx.x] += red[threadIdx.x + o];
        __syncthreads();
    }
    if (threadIdx.x == 0) pden[blockIdx.x] = red[0];
}

__global__ void finalize_k(const float* __restrict__ pnum, int nn,
                           const float* __restrict__ pden, int nd,
                           float* __restrict__ out) {
    if (blockIdx.x == 0 && threadIdx.x == 0) {
        float num = 0.f, den = 0.f;
        for (int i = 0; i < nn; ++i) num += pnum[i];
        for (int i = 0; i < nd; ++i) den += pden[i];
        out[0] = (den > 0.f) ? num / den : 0.f;
    }
}

// ---------------- host-side orchestration ----------------
extern "C" void kernel_launch(void* const* d_in, const int* in_sizes, int n_in,
                              void* d_out, int out_size, void* d_ws, size_t ws_size,
                              hipStream_t stream) {
    const float* preds = (const float*)d_in[0];   // [8,21,80,80]
    const float* lowf  = (const float*)d_in[1];   // [8,3,80,80]  (resize is identity)
    const float* highf = (const float*)d_in[2];   // [8,256,80,80]
    const int*   roi   = (const int*)  d_in[3];   // [8,80,80]    (resize is identity)
    float* out = (float*)d_out;

    char* wsp = (char*)d_ws;
    auto alloc = [&](size_t bytes) -> void* {
        void* p = (void*)wsp;
        wsp += (bytes + 255) & ~(size_t)255;
        return p;
    };
    const size_t BCV = (size_t)B_ * C_ * V_;
    const size_t BSV = (size_t)B_ * (C_ + 1) * V_;
    const size_t BV  = (size_t)B_ * V_;

    float* prob = (float*)alloc(BCV * 4);
    float* AS1  = (float*)alloc(BCV * 4);
    float* AS2  = (float*)alloc(BCV * 4);
    float* A    = (float*)alloc(BSV * 4);
    int* si_l = (int*)alloc(BV * 4);
    int* sp_l = (int*)alloc(BV * 4);
    int* si_h = (int*)alloc(BV * 4);
    int* sp_h = (int*)alloc(BV * 4);
    float* w_l = (float*)alloc(BV * 4);
    float* w_h = (float*)alloc(BV * 4);
    int* ufp  = (int*)alloc(BV * 4);
    int* mstu = (int*)alloc((size_t)B_ * (V_ - 1) * 4);
    int* mstv = (int*)alloc((size_t)B_ * (V_ - 1) * 4);
    int* deg  = (int*)alloc(BV * 4);
    int* off  = (int*)alloc((size_t)B_ * (V_ + 1) * 4);
    int* adj  = (int*)alloc((size_t)B_ * 2 * (V_ - 1) * 4);
    int* vis  = (int*)alloc(BV * 4);
    const int NUM_WAVES = 2048;   // 256 blocks * 256 threads / 32
    const int DEN_BLKS  = 64;
    float* pnum = (float*)alloc(NUM_WAVES * 4);
    float* pden = (float*)alloc(DEN_BLKS * 4);

    const int T = 256;
    // sps + sis + ws + CPG*Sc + CPG*Ac = (3 + 2*CPG) * V * 4 = 281600 bytes
    const size_t LDS_FILTER = (size_t)(3 + 2 * CPG) * V_ * 4;

    // 1) softmax
    softmax_k<<<(B_ * V_ + T - 1) / T, T, 0, stream>>>(preds, prob);
    // 2) MST + BFS trees (low then high; same scratch, stream-serialized)
    build_tree_k<<<B_, 1024, NSORT * sizeof(u64), stream>>>(lowf, 3, si_l, sp_l,
                                                            ufp, mstu, mstv, deg, off, adj, vis);
    build_tree_k<<<B_, 1024, NSORT * sizeof(u64), stream>>>(highf, 256, si_h, sp_h,
                                                            ufp, mstu, mstv, deg, off, adj, vis);
    // 3) edge weights along each tree
    compute_w_k<<<(B_ * V_ + T - 1) / T, T, 0, stream>>>(lowf, 3, si_l, sp_l, w_l);
    compute_w_k<<<(B_ * V_ + T - 1) / T, T, 0, stream>>>(highf, 256, si_h, sp_h, w_h);
    // 4) tree filter #1: prob filtered on low-feature tree (fused up+down, LDS-resident)
    dim3 gridF(B_, NGRP);
    tree_filter_k<<<gridF, 128, LDS_FILTER, stream>>>(prob, si_l, sp_l, w_l, A);
    normalize_k<<<(int)((BCV + T - 1) / T), T, 0, stream>>>(A, si_l, AS1);
    // 5) tree filter #2: AS1 filtered on high-feature tree
    tree_filter_k<<<gridF, 128, LDS_FILTER, stream>>>(AS1, si_h, sp_h, w_h, A);
    normalize_k<<<(int)((BCV + T - 1) / T), T, 0, stream>>>(A, si_h, AS2);
    // 6) loss via WMMA reduction + deterministic finalize
    loss_num_k<<<256, 256, 0, stream>>>(prob, AS2, roi, pnum);
    loss_den_k<<<DEN_BLKS, 256, 0, stream>>>(roi, pden);
    finalize_k<<<1, 32, 0, stream>>>(pnum, NUM_WAVES, pden, DEN_BLKS, out);
    (void)in_sizes; (void)n_in; (void)out_size; (void)ws_size;
}
